// MaskedAttention_24240795419222
// MI455X (gfx1250) — matrix operand order repaired and checked by reading, not compiled, not run on hardware
//
#include <hip/hip_runtime.h>
#include <hip/hip_bf16.h>

// ---------------------------------------------------------------------------
// Types for CDNA5 WMMA (wave32): bf16 A/B fragments (16 x __bf16 = 8 VGPRs),
// fp32 C/D accumulator (8 x float = 8 VGPRs).
// ---------------------------------------------------------------------------
typedef __attribute__((ext_vector_type(16))) __bf16 v16bf;
typedef __attribute__((ext_vector_type(8)))  __bf16 v8bf;
typedef __attribute__((ext_vector_type(8)))  float  v8f;

#define DIM 1024
#define SEQ 2048
#define NB  4
#define NROWS (NB * SEQ)          // 8192 total (b,t) rows

// fp32 -> bf16 round-to-nearest-even
__device__ __forceinline__ unsigned short f2bf(float f) {
    unsigned u = __float_as_uint(f);
    unsigned r = ((u >> 16) & 1u) + 0x7FFFu;
    return (unsigned short)((u + r) >> 16);
}
__device__ __forceinline__ float bf2f(unsigned short h) {
    return __uint_as_float(((unsigned)h) << 16);
}

// A fragment: 16x32 (MxK) bf16, row-major source with row stride `stride`
// (elements). Lane l: row = l&15; lanes 0-15 take K chunks [0..7],[16..23],
// lanes 16-31 take [8..15],[24..31]  (ISA 7.12.2, 16-bit A layout).
__device__ __forceinline__ v16bf load_a_frag(const unsigned short* base,
                                             int stride, int lane) {
    const unsigned short* p = base + (size_t)(lane & 15) * stride + ((lane & 16) ? 8 : 0);
    v16bf a;
    ((v8bf*)&a)[0] = *(const v8bf*)(p);
    ((v8bf*)&a)[1] = *(const v8bf*)(p + 16);
    return a;
}

// B fragment: 32x16 (KxN) bf16 from an "n-major" array: column n is contiguous
// along K with stride `stride` elements between columns. Lane l: n = l&15;
// lanes 0-15 hold K [0..7],[16..23], lanes 16-31 hold K [8..15],[24..31].
__device__ __forceinline__ v16bf load_b_frag(const unsigned short* base,
                                             int stride, int lane) {
    const unsigned short* p = base + (size_t)(lane & 15) * stride + ((lane & 16) ? 8 : 0);
    v16bf b;
    ((v8bf*)&b)[0] = *(const v8bf*)(p);
    ((v8bf*)&b)[1] = *(const v8bf*)(p + 16);
    return b;
}

__device__ __forceinline__ v8f wmma_bf16(v16bf a, v16bf b, v8f c) {
    return __builtin_amdgcn_wmma_f32_16x16x32_bf16(
        /*neg_a=*/false, a, /*neg_b=*/false, b,
        /*c_mod=*/(short)0, c, /*reuse_a=*/false, /*reuse_b=*/false);
}

// ---------------------------------------------------------------------------
// Kernel 1: X fp32 -> bf16 (4 elems / thread)
// ---------------------------------------------------------------------------
__global__ __launch_bounds__(256) void k_cvt_x(const float* __restrict__ x,
                                               unsigned short* __restrict__ xb) {
    int i = (blockIdx.x * blockDim.x + threadIdx.x) * 4;
    float4 v = *(const float4*)(x + i);
    ushort4 o;
    o.x = f2bf(v.x); o.y = f2bf(v.y); o.z = f2bf(v.z); o.w = f2bf(v.w);
    *(ushort4*)(xb + i) = o;
}

// ---------------------------------------------------------------------------
// Kernel 2: W fp32 [Din][Dout] -> bf16 transposed "n-major" [3][Dout][Din].
// 1/sqrt(Din) scale folded into Wq.
// ---------------------------------------------------------------------------
__global__ __launch_bounds__(256) void k_cvt_w(const float* __restrict__ wq,
                                               const float* __restrict__ wk,
                                               const float* __restrict__ wv,
                                               unsigned short* __restrict__ wt,
                                               float qscale) {
    int idx = blockIdx.x * blockDim.x + threadIdx.x;   // 3*1024*1024 threads
    int w = idx >> 20;
    int r = idx & 0xFFFFF;
    int d = r >> 10;
    int e = r & 1023;
    const float* src = (w == 0) ? wq : ((w == 1) ? wk : wv);
    float f = src[(size_t)d * DIM + e];
    if (w == 0) f *= qscale;
    wt[(size_t)w * DIM * DIM + (size_t)e * DIM + d] = f2bf(f);
}

// ---------------------------------------------------------------------------
// Kernel 3: fused QKV GEMM.  Each wave: 16 rows x 64 cols of one of Q/K/V.
// Q,K stored row-major bf16 [8192][1024]; V stored transposed Vt[b][e][t].
// Waves: 3 * 512 rowtiles * 16 colgroups = 24576.
// ---------------------------------------------------------------------------
__global__ __launch_bounds__(256) void k_qkv(const unsigned short* __restrict__ xb,
                                             const unsigned short* __restrict__ wt,
                                             unsigned short* __restrict__ qb,
                                             unsigned short* __restrict__ kb,
                                             unsigned short* __restrict__ vt) {
    int wid  = (blockIdx.x * blockDim.x + threadIdx.x) >> 5;
    int lane = threadIdx.x & 31;
    int w   = wid / (512 * 16);            // 0=Q 1=K 2=V
    int rem = wid % (512 * 16);
    int rt  = rem / 16;                    // row tile (16 rows)
    int cg  = rem % 16;                    // col group (64 cols)

    const unsigned short* abase = xb + (size_t)rt * 16 * DIM;
    const unsigned short* bbase = wt + (size_t)w * DIM * DIM + (size_t)cg * 64 * DIM;

    v8f acc[4] = {};
    for (int k0 = 0; k0 < DIM; k0 += 32) {
        __builtin_prefetch(abase + k0 + 64, 0, 3);
        v16bf a = load_a_frag(abase + k0, DIM, lane);
#pragma unroll
        for (int nt = 0; nt < 4; ++nt) {
            v16bf b = load_b_frag(bbase + (size_t)nt * 16 * DIM + k0, DIM, lane);
            acc[nt] = wmma_bf16(a, b, acc[nt]);
        }
    }

    int rowadd = (lane & 16) ? 8 : 0;
    int nlo    = lane & 15;
#pragma unroll
    for (int nt = 0; nt < 4; ++nt) {
        int n = cg * 64 + nt * 16 + nlo;
#pragma unroll
        for (int r = 0; r < 8; ++r) {
            int row = rt * 16 + r + rowadd;           // global (b,t) row
            unsigned short h = f2bf(acc[nt][r]);
            if (w == 0) {
                qb[(size_t)row * DIM + n] = h;
            } else if (w == 1) {
                kb[(size_t)row * DIM + n] = h;
            } else {
                int b = row >> 11, t = row & (SEQ - 1);
                vt[(size_t)b * DIM * SEQ + (size_t)n * SEQ + t] = h;
            }
        }
    }
}

// ---------------------------------------------------------------------------
// Kernel 4: scores S = Q K^T (scale already folded into Q). Causal: only
// k-tile groups with 4g <= qt are computed. Output bf16, row stride 2048.
// Waves: 4 * 128 qtiles * 32 kgroups = 16384 (about half early-exit).
// ---------------------------------------------------------------------------
__global__ __launch_bounds__(256) void k_scores(const unsigned short* __restrict__ qb,
                                                const unsigned short* __restrict__ kb,
                                                unsigned short* __restrict__ s) {
    int wid  = (blockIdx.x * blockDim.x + threadIdx.x) >> 5;
    int lane = threadIdx.x & 31;
    int b   = wid / (128 * 32);
    int rem = wid % (128 * 32);
    int qt  = rem / 32;
    int g   = rem % 32;
    if (4 * g > qt) return;                 // fully above the diagonal

    const unsigned short* qbase = qb + (size_t)(b * SEQ + qt * 16) * DIM;
    const unsigned short* kbase = kb + (size_t)(b * SEQ + g * 64) * DIM;

    v8f acc[4] = {};
    for (int k0 = 0; k0 < DIM; k0 += 32) {
        __builtin_prefetch(qbase + k0 + 64, 0, 3);
        v16bf a = load_a_frag(qbase + k0, DIM, lane);
#pragma unroll
        for (int nt = 0; nt < 4; ++nt) {
            v16bf bb = load_b_frag(kbase + (size_t)nt * 16 * DIM + k0, DIM, lane);
            acc[nt] = wmma_bf16(a, bb, acc[nt]);
        }
    }

    int rowadd = (lane & 16) ? 8 : 0;
    int nlo    = lane & 15;
#pragma unroll
    for (int nt = 0; nt < 4; ++nt) {
        int col = g * 64 + nt * 16 + nlo;
#pragma unroll
        for (int r = 0; r < 8; ++r) {
            int row = b * SEQ + qt * 16 + r + rowadd;
            s[(size_t)row * SEQ + col] = f2bf(acc[nt][r]);
        }
    }
}

// ---------------------------------------------------------------------------
// Kernel 5: causal row softmax, in place on bf16 score rows. One 256-thread
// block per row; valid columns = [0, t]; columns (t, 2047] zeroed so P.V
// needs no masking.
// ---------------------------------------------------------------------------
__global__ __launch_bounds__(256) void k_softmax(unsigned short* __restrict__ s) {
    int row = blockIdx.x;                  // 0..8191
    int t   = row & (SEQ - 1);
    int n   = t + 1;
    unsigned short* srow = s + (size_t)row * SEQ;
    __shared__ float buf[SEQ];
    __shared__ float red[256];
    int tid = threadIdx.x;

    float mx = -3.4e38f;
    for (int j = tid; j < n; j += 256) {
        float v = bf2f(srow[j]);
        buf[j] = v;
        mx = fmaxf(mx, v);
    }
    red[tid] = mx;
    __syncthreads();
    for (int st = 128; st > 0; st >>= 1) {
        if (tid < st) red[tid] = fmaxf(red[tid], red[tid + st]);
        __syncthreads();
    }
    mx = red[0];
    __syncthreads();

    float sum = 0.f;
    for (int j = tid; j < n; j += 256) {
        float e = __expf(buf[j] - mx);
        buf[j] = e;
        sum += e;
    }
    red[tid] = sum;
    __syncthreads();
    for (int st = 128; st > 0; st >>= 1) {
        if (tid < st) red[tid] += red[tid + st];
        __syncthreads();
    }
    float inv = 1.0f / red[0];

    for (int j = tid; j < n; j += 256)
        srow[j] = f2bf(buf[j] * inv);
    for (int j = n + tid; j < SEQ; j += 256)
        srow[j] = 0;
}

// ---------------------------------------------------------------------------
// Kernel 6: O = P V.  A = P rows (bf16, stride 2048); B = Vt (n-major over t).
// k-loop truncated at causal bound. Output fp32 straight to d_out.
// Waves: 4 * 128 rowtiles * 16 colgroups = 8192.
// ---------------------------------------------------------------------------
__global__ __launch_bounds__(256) void k_out(const unsigned short* __restrict__ p,
                                             const unsigned short* __restrict__ vt,
                                             float* __restrict__ out) {
    int wid  = (blockIdx.x * blockDim.x + threadIdx.x) >> 5;
    int lane = threadIdx.x & 31;
    int b   = wid / (128 * 16);
    int rem = wid % (128 * 16);
    int rt  = rem / 16;                    // row tile within batch
    int eg  = rem % 16;                    // 64-wide column group over e

    const unsigned short* pbase = p  + (size_t)(b * SEQ + rt * 16) * SEQ;
    const unsigned short* vbase = vt + (size_t)b * DIM * SEQ + (size_t)eg * 64 * SEQ;

    int iters = rt / 2 + 1;                // ceil((rt*16+16)/32)
    v8f acc[4] = {};
    for (int it = 0; it < iters; ++it) {
        int k0 = it * 32;
        __builtin_prefetch(pbase + k0 + 64, 0, 3);
        v16bf a = load_a_frag(pbase + k0, SEQ, lane);
#pragma unroll
        for (int nt = 0; nt < 4; ++nt) {
            v16bf bb = load_b_frag(vbase + (size_t)nt * 16 * SEQ + k0, SEQ, lane);
            acc[nt] = wmma_bf16(a, bb, acc[nt]);
        }
    }

    int rowadd = (lane & 16) ? 8 : 0;
    int nlo    = lane & 15;
#pragma unroll
    for (int nt = 0; nt < 4; ++nt) {
        int e = eg * 64 + nt * 16 + nlo;
#pragma unroll
        for (int r = 0; r < 8; ++r) {
            int row = b * SEQ + rt * 16 + r + rowadd;
            out[(size_t)row * DIM + e] = acc[nt][r];
        }
    }
}

// ---------------------------------------------------------------------------
// Host-side launcher
// ---------------------------------------------------------------------------
extern "C" void kernel_launch(void* const* d_in, const int* in_sizes, int n_in,
                              void* d_out, int out_size, void* d_ws, size_t ws_size,
                              hipStream_t stream) {
    const float* X  = (const float*)d_in[0];
    const float* Wq = (const float*)d_in[1];
    const float* Wk = (const float*)d_in[2];
    const float* Wv = (const float*)d_in[3];
    float* out = (float*)d_out;

    char* ws = (char*)d_ws;
    size_t off = 0;
    unsigned short* Xb = (unsigned short*)(ws + off); off += (size_t)NROWS * DIM * 2;       // 16 MB
    unsigned short* Wt = (unsigned short*)(ws + off); off += (size_t)3 * DIM * DIM * 2;     //  6 MB
    unsigned short* Qb = (unsigned short*)(ws + off); off += (size_t)NROWS * DIM * 2;       // 16 MB
    unsigned short* Kb = (unsigned short*)(ws + off); off += (size_t)NROWS * DIM * 2;       // 16 MB
    unsigned short* Vt = (unsigned short*)(ws + off); off += (size_t)NB * DIM * SEQ * 2;    // 16 MB
    unsigned short* S  = (unsigned short*)(ws + off); off += (size_t)NROWS * SEQ * 2;       // 32 MB

    const float qscale = 1.0f / 32.0f;     // 1/sqrt(1024)

    // 1) X -> bf16
    k_cvt_x<<<(NROWS * DIM) / (256 * 4), 256, 0, stream>>>(X, Xb);
    // 2) W -> bf16 transposed (+scale on Wq)
    k_cvt_w<<<(3 * DIM * DIM) / 256, 256, 0, stream>>>(Wq, Wk, Wv, Wt, qscale);
    // 3) QKV projection (24576 waves, 8 waves/block)
    k_qkv<<<(3 * 512 * 16) / 8, 256, 0, stream>>>(Xb, Wt, Qb, Kb, Vt);
    // 4) causal scores (16384 waves)
    k_scores<<<(NB * 128 * 32) / 8, 256, 0, stream>>>(Qb, Kb, S);
    // 5) row softmax in place
    k_softmax<<<NROWS, 256, 0, stream>>>(S);
    // 6) O = P V -> d_out (8192 waves)
    k_out<<<(NB * 128 * 16) / 8, 256, 0, stream>>>(S, Vt, out);
}
